// DCTSGCNLayer_24180665876674
// MI455X (gfx1250) — matrix operand order repaired and checked by reading, not compiled, not run on hardware
//
#include <hip/hip_runtime.h>

// ---------------------------------------------------------------------------
// DCT-SGCN hetero-GNN layer stack for MI455X (gfx1250).
// Scatter-bound workload: gathers/scatters via float4 + f32 global atomics.
// Dense [N,128]@[128,128] GEMMs on v_wmma_f32_16x16x32_bf16, fp32 accum.
// Weights pre-transposed+cast to bf16 once, staged per-block into LDS with
// global_load_async_to_lds_b128 (ASYNCcnt) — the CDNA5 async-LDS path.
// ---------------------------------------------------------------------------

#define HDIM 128
#define NPAP 200000
#define NAUT 100000
#define NSNP 20000
#define EWR  400000
#define ECI  600000
#define EIN  200000
#define ESN  100000
#define KLAY 3

typedef __attribute__((ext_vector_type(16))) __bf16 v16bf;
typedef __attribute__((ext_vector_type(8)))  float  v8f;

__device__ __forceinline__ unsigned short f2bf(float f) {
    unsigned int u = __float_as_uint(f);
    unsigned int r = (u + 0x7FFFu + ((u >> 16) & 1u)) >> 16;  // RNE
    return (unsigned short)r;
}

// degree[dst[e]] += 1
__global__ void k_deg(const int* __restrict__ dst, int E, float* __restrict__ deg) {
    int e = blockIdx.x * blockDim.x + threadIdx.x;
    if (e < E) atomicAdd(&deg[dst[e]], 1.0f);
}

// acc[dst[e], :] += feat[src[e], :] * (ew ? ew[e] : 1).  32 lanes/edge, 4 floats/lane.
__global__ void k_scatter(const float* __restrict__ feat, const int* __restrict__ src,
                          const int* __restrict__ dst, const float* __restrict__ ew,
                          int E, float* __restrict__ acc) {
    long t = (long)blockIdx.x * blockDim.x + threadIdx.x;
    int e = (int)(t >> 5);
    if (e >= E) return;
    int c = ((int)t & 31) << 2;
    const float4 v = *(const float4*)(feat + (long)src[e] * HDIM + c);
    float w = ew ? ew[e] : 1.0f;
    float* d = acc + (long)dst[e] * HDIM + c;
    atomicAdd(d + 0, v.x * w);
    atomicAdd(d + 1, v.y * w);
    atomicAdd(d + 2, v.z * w);
    atomicAdd(d + 3, v.w * w);
}

// out[i] = bf16( maybe_relu( in[i] / max(deg[row],1) ) )
__global__ void k_cast(const float* __restrict__ in, const float* __restrict__ deg,
                       int relu, unsigned short* __restrict__ out, long n) {
    long i = (long)blockIdx.x * blockDim.x + threadIdx.x;
    if (i >= n) return;
    float v = in[i];
    if (deg) v = v / fmaxf(deg[i >> 7], 1.0f);
    if (relu) v = fmaxf(v, 0.0f);
    out[i] = f2bf(v);
}

// One-time weight prep: fp32 row-major [mat][k][n] -> bf16 transposed [mat][n][k]
__global__ void k_wprep(const float* __restrict__ src, unsigned short* __restrict__ dst,
                        long nElem) {
    long i = (long)blockIdx.x * blockDim.x + threadIdx.x;
    if (i >= nElem) return;
    long mat = i >> 14;
    int r = (int)(i & 16383);
    int k = r >> 7, n = r & 127;
    dst[(mat << 14) + (n << 7) + k] = f2bf(src[i]);
}

// out[N,128] = (beta ? out : 0) + A_bf16[N,128] @ W[128,128] + bias
// Wt is pre-transposed bf16 [n][k].  One block = 256 threads = 8 waves; wave w
// computes rows [blk*128 + 16w, +16) x all 128 cols.
__global__ __launch_bounds__(256)
void k_gemm128(const unsigned short* __restrict__ A, const unsigned short* __restrict__ Wt,
               const float* __restrict__ bias, float* __restrict__ out,
               int N, int beta) {
    __shared__ unsigned short wt[HDIM * HDIM];  // bf16 weights [n][k], 32 KB

    // Stage weights global->LDS with the CDNA5 async-LDS copy path.
    // 256 threads x 8 x b128 = 32 KB; tracked on ASYNCcnt.
    {
        unsigned ldsAddr = (unsigned)(unsigned long long)(void*)&wt[0]
                         + (unsigned)threadIdx.x * 128u;
        const unsigned short* g = Wt + threadIdx.x * 64;
#pragma unroll
        for (int j = 0; j < 8; ++j) {
            asm volatile("global_load_async_to_lds_b128 %0, %1, off"
                         :: "v"(ldsAddr + (unsigned)(j * 16)), "v"(g + j * 8)
                         : "memory");
        }
        asm volatile("s_wait_asynccnt 0" ::: "memory");
    }
    __syncthreads();

    int wave = threadIdx.x >> 5;
    int lane = threadIdx.x & 31;
    int g    = lane >> 4;      // lane group selects K-half of fragments
    int lr   = lane & 15;      // row (A) / col (B,D) within tile
    int rowBase = blockIdx.x * 128 + wave * 16;

    int lrow = rowBase + lr;           // clamp loads so EXEC stays all-ones for WMMA
    if (lrow > N - 1) lrow = N - 1;
    const unsigned short* arow = A + (long)lrow * HDIM;

    v8f acc[8];
#pragma unroll
    for (int t = 0; t < 8; ++t) acc[t] = (v8f){0.f, 0.f, 0.f, 0.f, 0.f, 0.f, 0.f, 0.f};

    union Frag { uint4 q[2]; v16bf v; };

#pragma unroll
    for (int kb = 0; kb < HDIM; kb += 32) {
        // A 16x32 bf16 fragment: lane row = lr; elems 0..7 -> K=kb+8g+0..7,
        // elems 8..15 -> K=kb+16+8g+0..7  (ISA 16-bit A layout)
        Frag a;
        a.q[0] = *(const uint4*)(arow + kb + g * 8);
        a.q[1] = *(const uint4*)(arow + kb + 16 + g * 8);

        // Load all 8 B fragments first (distinct regs) so the scheduler can
        // clause the 16 ds_load_b128s and then run 8 WMMAs back-to-back.
        Frag b[8];
#pragma unroll
        for (int nt = 0; nt < 8; ++nt) {
            const unsigned short* bp = &wt[(nt * 16 + lr) * HDIM + kb + g * 8];
            b[nt].q[0] = *(const uint4*)(bp);
            b[nt].q[1] = *(const uint4*)(bp + 16);
        }
#pragma unroll
        for (int nt = 0; nt < 8; ++nt) {
            acc[nt] = __builtin_amdgcn_wmma_f32_16x16x32_bf16(
                false, a.v, false, b[nt].v, (short)0, acc[nt], false, false);
        }
    }

    // D layout: lane 0-15 -> col=lane, M=r; lane 16-31 -> col=lane-16, M=8+r
#pragma unroll
    for (int nt = 0; nt < 8; ++nt) {
        int col = nt * 16 + lr;
        float bv = bias ? bias[col] : 0.0f;
#pragma unroll
        for (int r = 0; r < 8; ++r) {
            int row = rowBase + g * 8 + r;
            if (row < N) {
                long o = (long)row * HDIM + col;
                float v = acc[nt][r] + bv;
                if (beta) v += out[o];
                out[o] = v;
            }
        }
    }
}

// ---------------------------------------------------------------------------

extern "C" void kernel_launch(void* const* d_in, const int* in_sizes, int n_in,
                              void* d_out, int out_size, void* d_ws, size_t ws_size,
                              hipStream_t stream) {
    (void)in_sizes; (void)n_in; (void)out_size; (void)ws_size;

    const float* xP = (const float*)d_in[0];
    const float* xA = (const float*)d_in[1];
    const float* xS = (const float*)d_in[2];
    const int* wr_s = (const int*)d_in[3];  const int* wr_d = (const int*)d_in[4];
    const int* ci_s = (const int*)d_in[5];  const int* ci_d = (const int*)d_in[6];
    const int* in_s = (const int*)d_in[7];  const int* in_d = (const int*)d_in[8];
    const int* sn_s = (const int*)d_in[9];  const int* sn_d = (const int*)d_in[10];
    const float* wsn   = (const float*)d_in[11];
    const float* convW = (const float*)d_in[12];
    const float* convB = (const float*)d_in[13];
    const float* skipW = (const float*)d_in[14];
    const float* skipB = (const float*)d_in[15];
    const float* trnW  = (const float*)d_in[16];
    const float* trnB  = (const float*)d_in[17];
    const float* ccW   = (const float*)d_in[18];
    const float* ccB   = (const float*)d_in[19];

    const long P = (long)NPAP * HDIM, AE = (long)NAUT * HDIM, SE = (long)NSNP * HDIM;
    const long MAT = 16384;  // elements in a 128x128 block

    // ---- workspace carve-out (~825 MB) ----
    char* wp = (char*)d_ws;
    auto alloc = [&](size_t b) -> void* {
        void* p = (void*)wp; wp += (b + 255) & ~(size_t)255; return p;
    };
    float* maccP = (float*)alloc(P * 4);   // msg accumulators (aliased as bwd-trans out)
    float* maccA = (float*)alloc(AE * 4);
    float* maccS = (float*)alloc(SE * 4);
    float* tP = (float*)alloc(P * 4);      // p2/a2/s2 fp32
    float* tA = (float*)alloc(AE * 4);
    float* tS = (float*)alloc(SE * 4);
    float* fP = (float*)alloc(P * 4);      // forward trans outputs
    float* fA = (float*)alloc(AE * 4);
    float* fS = (float*)alloc(SE * 4);
    float* nxP = (float*)alloc(P * 4);     // next-layer features
    float* nxA = (float*)alloc(AE * 4);
    float* nxS = (float*)alloc(SE * 4);
    unsigned short* bfP  = (unsigned short*)alloc(P * 2);   // bf16 of cur features
    unsigned short* bfA  = (unsigned short*)alloc(AE * 2);
    unsigned short* bfS  = (unsigned short*)alloc(SE * 2);
    unsigned short* bmP  = (unsigned short*)alloc(P * 2);   // bf16 msg / relu temps
    unsigned short* bmA  = (unsigned short*)alloc(AE * 2);
    unsigned short* bmS  = (unsigned short*)alloc(SE * 2);
    float* degPwr  = (float*)alloc((size_t)NPAP * 4);
    float* degPci  = (float*)alloc((size_t)NPAP * 4);
    float* degSin  = (float*)alloc((size_t)NSNP * 4);
    float* degAwrb = (float*)alloc((size_t)NAUT * 4);
    float* degPcib = (float*)alloc((size_t)NPAP * 4);
    float* degPinb = (float*)alloc((size_t)NPAP * 4);
    unsigned short* convT = (unsigned short*)alloc((size_t)(KLAY * 2 * 4) * MAT * 2);
    unsigned short* skipT = (unsigned short*)alloc((size_t)(KLAY * 2 * 2) * MAT * 2);
    unsigned short* trnT  = (unsigned short*)alloc((size_t)(KLAY * 2 * 3) * MAT * 2);
    unsigned short* ccT   = (unsigned short*)alloc((size_t)(KLAY * 3 * 2) * MAT * 2);
    float* bP = maccP; float* bA = maccA; float* bS = maccS;  // backward trans outputs

    // ---- launch helpers ----
    auto zero = [&](void* p, size_t bytes) { hipMemsetAsync(p, 0, bytes, stream); };
    auto deg = [&](const int* d, int E, float* dg) {
        hipLaunchKernelGGL(k_deg, dim3((E + 255) / 256), dim3(256), 0, stream, d, E, dg);
    };
    auto scat = [&](const float* feat, const int* s, const int* d, const float* ew,
                    int E, float* acc) {
        long T = (long)E * 32;
        hipLaunchKernelGGL(k_scatter, dim3((unsigned)((T + 255) / 256)), dim3(256), 0,
                           stream, feat, s, d, ew, E, acc);
    };
    auto cast = [&](const float* in, const float* dgp, int relu,
                    unsigned short* out, long n) {
        hipLaunchKernelGGL(k_cast, dim3((unsigned)((n + 255) / 256)), dim3(256), 0,
                           stream, in, dgp, relu, out, n);
    };
    auto wprep = [&](const float* src, unsigned short* dst, long nMat) {
        long n = nMat * MAT;
        hipLaunchKernelGGL(k_wprep, dim3((unsigned)((n + 255) / 256)), dim3(256), 0,
                           stream, src, dst, n);
    };
    auto gemm = [&](const unsigned short* A, const unsigned short* Wt, const float* bias,
                    float* out, int N, int beta) {
        hipLaunchKernelGGL(k_gemm128, dim3((N + 127) / 128), dim3(256), 0, stream,
                           A, Wt, bias, out, N, beta);
    };

    // transposed-bf16 weight slice helpers
    auto cw = [&](int i, int d, int j) { return convT + (((long)(i * 2 + d)) * 4 + j) * MAT; };
    auto cb = [&](int i, int d, int j) { return convB + (((long)(i * 2 + d)) * 4 + j) * HDIM; };
    auto sw = [&](int i, int d, int j) { return skipT + (((long)(i * 2 + d)) * 2 + j) * MAT; };
    auto sb = [&](int i, int d, int j) { return skipB + (((long)(i * 2 + d)) * 2 + j) * HDIM; };
    auto tw = [&](int i, int d, int j) { return trnT + (((long)(i * 2 + d)) * 3 + j) * MAT; };
    auto tb = [&](int i, int d, int j) { return trnB + (((long)(i * 2 + d)) * 3 + j) * HDIM; };
    auto qw = [&](int i, int t) { return ccT + ((long)(i * 3 + t)) * 2 * MAT; };
    auto qb = [&](int i, int t) { return ccB + ((long)(i * 3 + t)) * HDIM; };

    // ---- one-time weight transpose+cast (2.4 MB bf16 total) ----
    wprep(convW, convT, KLAY * 2 * 4);
    wprep(skipW, skipT, KLAY * 2 * 2);
    wprep(trnW,  trnT,  KLAY * 2 * 3);
    wprep(ccW,   ccT,   KLAY * 3 * 2);   // each [2H,H] = two contiguous 128x128 blocks

    // ---- degrees (edge topology is fixed across layers) ----
    zero(degPwr, (size_t)NPAP * 4);  deg(wr_d, EWR, degPwr);
    zero(degPci, (size_t)NPAP * 4);  deg(ci_d, ECI, degPci);
    zero(degSin, (size_t)NSNP * 4);  deg(in_d, EIN, degSin);
    zero(degAwrb, (size_t)NAUT * 4); deg(wr_s, EWR, degAwrb);
    zero(degPcib, (size_t)NPAP * 4); deg(ci_s, ECI, degPcib);
    zero(degPinb, (size_t)NPAP * 4); deg(in_s, EIN, degPinb);

    const float *curP = xP, *curA = xA, *curS = xS;

    for (int i = 0; i < KLAY; ++i) {
        // bf16 of current features (used by skip GEMMs both directions)
        cast(curP, nullptr, 0, bfP, P);
        cast(curA, nullptr, 0, bfA, AE);
        cast(curS, nullptr, 0, bfS, SE);

        // ---------- forward direction (d=0) ----------
        // msg_p = gconv(a, writes) + gconv(p, cites); p2 = p@sW0+sb0 + msg_p
        zero(maccP, P * 4); scat(curA, wr_s, wr_d, nullptr, EWR, maccP);
        cast(maccP, degPwr, 0, bmP, P);
        gemm(bmP, cw(i, 0, 0), cb(i, 0, 0), tP, NPAP, 0);
        zero(maccP, P * 4); scat(curP, ci_s, ci_d, nullptr, ECI, maccP);
        cast(maccP, degPci, 0, bmP, P);
        gemm(bmP, cw(i, 0, 1), cb(i, 0, 1), tP, NPAP, 1);
        gemm(bfP, sw(i, 0, 0), sb(i, 0, 0), tP, NPAP, 1);          // tP = p2
        // msg_s = gconv(p, in); s2 = s@sW1+sb1 + msg_s
        zero(maccS, SE * 4); scat(curP, in_s, in_d, nullptr, EIN, maccS);
        cast(maccS, degSin, 0, bmS, SE);
        gemm(bmS, cw(i, 0, 2), cb(i, 0, 2), tS, NSNP, 0);
        gemm(bfS, sw(i, 0, 1), sb(i, 0, 1), tS, NSNP, 1);          // tS = s2 (pre-wconv)
        // s2 = wconv(s2, snap fwd)
        zero(maccS, SE * 4); scat(tS, sn_s, sn_d, wsn, ESN, maccS);
        cast(maccS, nullptr, 0, bmS, SE);
        gemm(bmS, cw(i, 0, 3), cb(i, 0, 3), tS, NSNP, 0);          // tS = s2 final
        // relu + trans -> forward outputs
        cast(tP, nullptr, 1, bmP, P);  gemm(bmP, tw(i, 0, 0), tb(i, 0, 0), fP, NPAP, 0);
        cast(curA, nullptr, 1, bmA, AE); gemm(bmA, tw(i, 0, 1), tb(i, 0, 1), fA, NAUT, 0);
        cast(tS, nullptr, 1, bmS, SE); gemm(bmS, tw(i, 0, 2), tb(i, 0, 2), fS, NSNP, 0);

        // ---------- backward direction (d=1) ----------
        // msg_a = gconv(p, writes reversed); a2 = a@sW0+sb0 + msg_a
        zero(maccA, AE * 4); scat(curP, wr_d, wr_s, nullptr, EWR, maccA);
        cast(maccA, degAwrb, 0, bmA, AE);
        gemm(bmA, cw(i, 1, 0), cb(i, 1, 0), tA, NAUT, 0);
        gemm(bfA, sw(i, 1, 0), sb(i, 1, 0), tA, NAUT, 1);          // tA = a2
        // msg_p = gconv(p, cites rev) + gconv(s, in rev); p2 = p@sW1+sb1 + msg_p
        zero(maccP, P * 4); scat(curP, ci_d, ci_s, nullptr, ECI, maccP);
        cast(maccP, degPcib, 0, bmP, P);
        gemm(bmP, cw(i, 1, 1), cb(i, 1, 1), tP, NPAP, 0);
        zero(maccP, P * 4); scat(curS, in_d, in_s, nullptr, EIN, maccP);
        cast(maccP, degPinb, 0, bmP, P);
        gemm(bmP, cw(i, 1, 2), cb(i, 1, 2), tP, NPAP, 1);
        gemm(bfP, sw(i, 1, 1), sb(i, 1, 1), tP, NPAP, 1);          // tP = p2
        // s2 = wconv(s, snap reversed)
        zero(maccS, SE * 4); scat(curS, sn_d, sn_s, wsn, ESN, maccS);
        cast(maccS, nullptr, 0, bmS, SE);
        gemm(bmS, cw(i, 1, 3), cb(i, 1, 3), tS, NSNP, 0);          // tS = s2
        // relu + trans -> backward outputs (aliased on macc buffers, now free)
        cast(tP, nullptr, 1, bmP, P);  gemm(bmP, tw(i, 1, 0), tb(i, 1, 0), bP, NPAP, 0);
        cast(tA, nullptr, 1, bmA, AE); gemm(bmA, tw(i, 1, 1), tb(i, 1, 1), bA, NAUT, 0);
        cast(tS, nullptr, 1, bmS, SE); gemm(bmS, tw(i, 1, 2), tb(i, 1, 2), bS, NSNP, 0);

        // ---------- concat [fwd, bwd] @ W(2H->H) ----------
        float* oP = nxP; float* oA = nxA; float* oS = nxS;
        if (i == KLAY - 1) {
            oP = (float*)d_out;
            oA = (float*)d_out + P;
            oS = (float*)d_out + P + AE;
        }
        cast(fP, nullptr, 0, bfP, P);  cast(bP, nullptr, 0, bmP, P);
        gemm(bfP, qw(i, 0), qb(i, 0), oP, NPAP, 0);
        gemm(bmP, qw(i, 0) + MAT, nullptr, oP, NPAP, 1);
        cast(fA, nullptr, 0, bfA, AE); cast(bA, nullptr, 0, bmA, AE);
        gemm(bfA, qw(i, 1), qb(i, 1), oA, NAUT, 0);
        gemm(bmA, qw(i, 1) + MAT, nullptr, oA, NAUT, 1);
        cast(fS, nullptr, 0, bfS, SE); cast(bS, nullptr, 0, bmS, SE);
        gemm(bfS, qw(i, 2), qb(i, 2), oS, NSNP, 0);
        gemm(bmS, qw(i, 2) + MAT, nullptr, oS, NSNP, 1);

        curP = oP; curA = oA; curS = oS;
    }
}
